// BackboneModule_60584808677379
// MI455X (gfx1250) — matrix operand order repaired and checked by reading, not compiled, not run on hardware
//
#include <hip/hip_runtime.h>

typedef __attribute__((ext_vector_type(16))) _Float16 v16h;
typedef __attribute__((ext_vector_type(8)))  float    v8f;

#define B_ 2
#define BN_INV 0.999995000037499f   // 1/sqrt(1+1e-5)

// ---------------------------------------------------------------------------
// Weight repack: f32 [cin][cout] -> f16 WMMA-fragment-tiled:
//   Wp[((kb*tilesN + tn)*32 + lane)*16 + h]
// where n = tn*16 + (lane&15), k = kb*32 + (lane>=16 ? 16 : 0) + h.
// Each lane's B fragment becomes 32 contiguous bytes (2 x b128 loads).
// ---------------------------------------------------------------------------
__global__ void wpack_kernel(const float* __restrict__ W, _Float16* __restrict__ Wp,
                             int cin, int cinPad, int cout) {
  int i = blockIdx.x * blockDim.x + threadIdx.x;
  if (i >= cinPad * cout) return;
  int h = i & 15;
  int lane = (i >> 4) & 31;
  int t = i >> 9;                 // tile index: kb*tilesN + tn
  int tilesN = cout >> 4;
  int tn = t % tilesN, kb = t / tilesN;
  int n = tn * 16 + (lane & 15);
  int k = kb * 32 + ((lane >> 4) << 4) + h;
  Wp[i] = (k < cin) ? (_Float16)W[(size_t)k * cout + n] : (_Float16)0.f;
}

// ---------------------------------------------------------------------------
// Farthest point sampling: one block per batch. The 240KB point cloud is
// staged into LDS with global_load_async_to_lds_b128 (VGPR-bypassing DMA,
// tracked by ASYNCcnt), then 2047 argmax rounds run out of LDS.
// ---------------------------------------------------------------------------
__global__ void fps_kernel(const float* __restrict__ pts, float* __restrict__ dist,
                           int* __restrict__ idx1, int Np, int M) {
  extern __shared__ float spts[];           // Np*3 floats (dynamic LDS)
  __shared__ float sd[1024];
  __shared__ int   si[1024];
  __shared__ int   s_last;

  int b = blockIdx.x;
  const float* P = pts + (size_t)b * Np * 3;
  float* D = dist + (size_t)b * Np;
  int* out = idx1 + (size_t)b * M;
  int t = threadIdx.x, nt = blockDim.x;

  // Async HBM -> LDS copy, 16B per lane per issue (Np*3*4 is 16B-divisible)
  {
    int chunks = (Np * 3 * 4) >> 4;
    unsigned ldsBase = (unsigned)(size_t)spts;   // low 32 bits = LDS offset
    for (int c = t; c < chunks; c += nt) {
      unsigned long long ga = (unsigned long long)(const char*)P + ((unsigned long long)c << 4);
      unsigned la = ldsBase + ((unsigned)c << 4);
      asm volatile("global_load_async_to_lds_b128 %0, %1, off"
                   :: "v"(la), "v"(ga) : "memory");
    }
    asm volatile("s_wait_asynccnt 0x0" ::: "memory");
  }
  for (int i = t; i < Np; i += nt) D[i] = 1e10f;
  if (t == 0) { out[0] = 0; s_last = 0; }
  __syncthreads();

  for (int it = 1; it < M; ++it) {
    int last = s_last;
    float lx = spts[last * 3], ly = spts[last * 3 + 1], lz = spts[last * 3 + 2];
    float best = -1.f; int bi = 0;
    for (int i = t; i < Np; i += nt) {
      float dx = spts[i * 3] - lx, dy = spts[i * 3 + 1] - ly, dz = spts[i * 3 + 2] - lz;
      float d = dx * dx + dy * dy + dz * dz;
      float dd = fminf(D[i], d); D[i] = dd;
      if (dd > best) { best = dd; bi = i; }
    }
    sd[t] = best; si[t] = bi;
    __syncthreads();
    for (int s = nt >> 1; s > 0; s >>= 1) {
      if (t < s) {
        if (sd[t + s] > sd[t] || (sd[t + s] == sd[t] && si[t + s] < si[t])) {
          sd[t] = sd[t + s]; si[t] = si[t + s];
        }
      }
      __syncthreads();
    }
    if (t == 0) { out[it] = si[0]; s_last = si[0]; }
    __syncthreads();
  }
}

// ---------------------------------------------------------------------------
// p1[b][m][c] = points[b][idx1[b][m]][c]
// ---------------------------------------------------------------------------
__global__ void gather3_kernel(const float* __restrict__ pts, const int* __restrict__ idx1,
                               float* __restrict__ p1, int Np, int M) {
  int i = blockIdx.x * blockDim.x + threadIdx.x;
  if (i >= B_ * M * 3) return;
  int b = i / (M * 3); int r = i % (M * 3);
  int m = r / 3, c = r % 3;
  p1[i] = pts[((size_t)b * Np + idx1[b * M + m]) * 3 + c];
}

__global__ void slice_kernel(const float* __restrict__ src, float* __restrict__ dst,
                             int wSrc, int wDst) {
  int i = blockIdx.x * blockDim.x + threadIdx.x;
  if (i >= B_ * wDst) return;
  int b = i / wDst; int r = i % wDst;
  dst[i] = src[(size_t)b * wSrc + r];
}

// ---------------------------------------------------------------------------
// Ball query: one wave32 per ref point; ordered compaction via ballot
// ---------------------------------------------------------------------------
__global__ void ball_query_kernel(const float* __restrict__ pts, const float* __restrict__ refs,
                                  int* __restrict__ nidx, int Np, int M, int K, float r2) {
  int wave = blockIdx.x * (blockDim.x >> 5) + (threadIdx.x >> 5);
  int lane = threadIdx.x & 31;
  if (wave >= B_ * M) return;
  int b = wave / M, m = wave % M;
  const float* P = pts + (size_t)b * Np * 3;
  const float* R = refs + ((size_t)b * M + m) * 3;
  int* out = nidx + ((size_t)b * M + m) * K;
  float rx = R[0], ry = R[1], rz = R[2];
  int count = 0, firstIdx = 0;
  for (int base = 0; base < Np && count < K; base += 32) {
    int i = base + lane;
    bool in = false;
    if (i < Np) {
      float dx = P[i * 3] - rx, dy = P[i * 3 + 1] - ry, dz = P[i * 3 + 2] - rz;
      in = (dx * dx + dy * dy + dz * dz) < r2;
    }
    unsigned mask = __builtin_amdgcn_ballot_w32(in);
    if (mask) {
      if (count == 0) firstIdx = base + __builtin_ctz(mask);
      int off = __popc(mask & ((1u << lane) - 1u));
      if (in && (count + off) < K) out[count + off] = i;
      count += __popc(mask);
    }
  }
  int c = count < K ? count : K;
  int fill = (c == 0) ? 0 : firstIdx;
  for (int j = c + lane; j < K; j += 32) out[j] = fill;
}

// ---------------------------------------------------------------------------
// Gather + relative-coord + feature concat into padded f16 GEMM input
// ---------------------------------------------------------------------------
__global__ void gather_concat_kernel(const float* __restrict__ pts, const _Float16* __restrict__ feat,
                                     const float* __restrict__ refs, const int* __restrict__ nidx,
                                     _Float16* __restrict__ X, int Np, int M, int K, int C,
                                     int cinPad, float invR, long long rowsTotal) {
  long long row = (long long)blockIdx.x * blockDim.x + threadIdx.x;
  if (row >= rowsTotal) return;
  int k = (int)(row % K);
  long long bm = row / K;
  int b = (int)(bm / M);
  int pi = nidx[bm * K + k];
  const float* P = pts + ((size_t)b * Np + pi) * 3;
  const float* R = refs + (size_t)bm * 3;
  _Float16* x = X + (size_t)row * cinPad;
  x[0] = (_Float16)((P[0] - R[0]) * invR);
  x[1] = (_Float16)((P[1] - R[1]) * invR);
  x[2] = (_Float16)((P[2] - R[2]) * invR);
  if (C > 0) {
    const _Float16* F = feat + ((size_t)b * Np + pi) * C;
    for (int c = 0; c < C; ++c) x[3 + c] = F[c];
  }
  for (int c = 3 + C; c < cinPad; ++c) x[c] = (_Float16)0.f;
}

// ---------------------------------------------------------------------------
// WMMA GEMM + folded-BN + ReLU.  One wave owns a 16x64 output strip:
// one A fragment feeds 4 back-to-back v_wmma_f32_16x16x32_f16 ops.
// B fragments come from the packed layout: 2 x b128 per tile per k-block.
// ---------------------------------------------------------------------------
__global__ void wmma_mlp_kernel(const _Float16* __restrict__ X, const _Float16* __restrict__ Wp,
                                const float* __restrict__ gamma, const float* __restrict__ beta,
                                _Float16* __restrict__ Y, int rows, int cinPad, int cout) {
  int wid = blockIdx.x * (blockDim.x >> 5) + (threadIdx.x >> 5);
  int lane = threadIdx.x & 31;
  int tilesN = cout >> 4;
  int groupsN = tilesN >> 2;               // 4 column tiles per wave
  int totalW = (rows >> 4) * groupsN;
  if (wid >= totalW) return;
  int tm = wid / groupsN, tg = wid % groupsN;
  int laneHi = lane >> 4;
  int lm = lane & 15;
  int m0 = tm * 16 + lm;

  union AF { v16h v; uint4 q[2]; } a;
  union BF { v16h v; uint4 q[2]; } bf;
  v8f acc[4] = {};

  const _Float16* xrow = X + (size_t)m0 * cinPad + laneHi * 16;
  int nk = cinPad >> 5;
  for (int kb = 0; kb < nk; ++kb) {
    const _Float16* xp = xrow + kb * 32;
    a.q[0] = *(const uint4*)(xp);
    a.q[1] = *(const uint4*)(xp + 8);
    // lane's packed B fragments: 32B contiguous per tile; tiles stride 1KB
    const uint4* bp = (const uint4*)(Wp + (((size_t)kb * tilesN + tg * 4) * 32 + lane) * 16);
    if (kb + 1 < nk) __builtin_prefetch(bp + (size_t)tilesN * 64, 0, 1);
#pragma unroll
    for (int j = 0; j < 4; ++j) {
      bf.q[0] = bp[j * 64];
      bf.q[1] = bp[j * 64 + 1];
      acc[j] = __builtin_amdgcn_wmma_f32_16x16x32_f16(false, a.v, false, bf.v,
                                                      (short)0, acc[j], false, false);
    }
  }
  int mo = tm * 16 + laneHi * 8;           // C/D layout: VGPR r -> M = r (+8 hi lanes)
#pragma unroll
  for (int j = 0; j < 4; ++j) {
    int nc = (tg * 4 + j) * 16 + lm;
    float g = gamma[nc] * BN_INV;
    float bb = beta[nc];
#pragma unroll
    for (int r = 0; r < 8; ++r) {
      float y = fmaxf(acc[j][r] * g + bb, 0.f);
      Y[(size_t)(mo + r) * cout + nc] = (_Float16)y;
    }
  }
}

// ---------------------------------------------------------------------------
// Max-pool over the K neighbor dimension (ReLU outputs -> init 0 is exact)
// ---------------------------------------------------------------------------
__global__ void maxpool_kernel(const _Float16* __restrict__ Y, _Float16* __restrict__ out,
                               int K, int C, long long bmTotal) {
  long long i = (long long)blockIdx.x * blockDim.x + threadIdx.x;
  if (i >= bmTotal * C) return;
  long long bm = i / C; int c = (int)(i % C);
  const _Float16* y = Y + bm * (size_t)K * C + c;
  float m = 0.f;
  for (int j = 0; j < K; ++j) m = fmaxf(m, (float)y[(size_t)j * C]);
  out[(size_t)bm * C + c] = (_Float16)m;
}

// ---------------------------------------------------------------------------
// 3-NN search + inverse-distance weights
// ---------------------------------------------------------------------------
__global__ void knn3_kernel(const float* __restrict__ unk, const float* __restrict__ kn,
                            int* __restrict__ idx3, float* __restrict__ w3,
                            int Mu, int Mk, int total) {
  int i = blockIdx.x * blockDim.x + threadIdx.x;
  if (i >= total) return;
  int b = i / Mu;
  const float* U = unk + (size_t)i * 3;
  const float* P = kn + (size_t)b * Mk * 3;
  float ux = U[0], uy = U[1], uz = U[2];
  float d0 = 1e30f, d1 = 1e30f, d2v = 1e30f; int i0 = 0, i1 = 0, i2v = 0;
  for (int j = 0; j < Mk; ++j) {
    float dx = P[j * 3] - ux, dy = P[j * 3 + 1] - uy, dz = P[j * 3 + 2] - uz;
    float d = dx * dx + dy * dy + dz * dz;
    if (d < d0)      { d2v = d1; i2v = i1; d1 = d0; i1 = i0; d0 = d; i0 = j; }
    else if (d < d1) { d2v = d1; i2v = i1; d1 = d; i1 = j; }
    else if (d < d2v){ d2v = d; i2v = j; }
  }
  float w0 = 1.f / (d0 + 1e-8f), w1 = 1.f / (d1 + 1e-8f), w2 = 1.f / (d2v + 1e-8f);
  float s = w0 + w1 + w2;
  idx3[i * 3] = i0; idx3[i * 3 + 1] = i1; idx3[i * 3 + 2] = i2v;
  w3[i * 3] = w0 / s; w3[i * 3 + 1] = w1 / s; w3[i * 3 + 2] = w2 / s;
}

__global__ void interp_kernel(const _Float16* __restrict__ feat, const int* __restrict__ idx3,
                              const float* __restrict__ w3, _Float16* __restrict__ out,
                              int Mu, int Mk, int C) {
  long long i = (long long)blockIdx.x * blockDim.x + threadIdx.x;
  if (i >= (long long)B_ * Mu * C) return;
  long long bm = i / C; int c = (int)(i % C);
  int b = (int)(bm / Mu);
  const _Float16* F = feat + (size_t)b * Mk * C;
  float acc = 0.f;
#pragma unroll
  for (int j = 0; j < 3; ++j)
    acc += w3[bm * 3 + j] * (float)F[(size_t)idx3[bm * 3 + j] * C + c];
  out[i] = (_Float16)acc;
}

__global__ void concat_kernel(const _Float16* __restrict__ A, const _Float16* __restrict__ Bf,
                              _Float16* __restrict__ X, int C1, int C2, long long rowsTotal) {
  long long i = (long long)blockIdx.x * blockDim.x + threadIdx.x;
  int W = C1 + C2;
  if (i >= rowsTotal * W) return;
  long long row = i / W; int c = (int)(i % W);
  X[i] = (c < C1) ? A[row * C1 + c] : Bf[row * C2 + (c - C1)];
}

// ---------------------------------------------------------------------------
// Emit (idx2 as float, p2, f2) flat into d_out
// ---------------------------------------------------------------------------
__global__ void emit_kernel(const int* __restrict__ idx1, const float* __restrict__ p1,
                            const _Float16* __restrict__ f2, float* __restrict__ out,
                            int Mi, int M1, int C, int total) {
  int i = blockIdx.x * blockDim.x + threadIdx.x;
  if (i >= total) return;
  int sec1 = B_ * Mi;
  int sec2 = sec1 + B_ * Mi * 3;
  if (i < sec1) {
    int b = i / Mi, m = i % Mi;
    out[i] = (float)idx1[b * M1 + m];
  } else if (i < sec2) {
    int j = i - sec1; int b = j / (Mi * 3); int r = j % (Mi * 3);
    out[i] = p1[(size_t)b * M1 * 3 + r];
  } else {
    int j = i - sec2;
    out[i] = (float)f2[j];
  }
}

// ---------------------------------------------------------------------------
// Host orchestration
// ---------------------------------------------------------------------------
extern "C" void kernel_launch(void* const* d_in, const int* in_sizes, int n_in,
                              void* d_out, int out_size, void* d_ws, size_t ws_size,
                              hipStream_t stream) {
  (void)in_sizes; (void)n_in; (void)out_size; (void)ws_size;
  const float* points = (const float*)d_in[0];

  struct LP { const float* W; const float* beta; const float* gamma;
              int cin, cinPad, cout; _Float16* Wh; };
  static const int dims[6][4] = {
    {3, 64, 64, 128}, {131, 128, 128, 256}, {259, 128, 128, 256},
    {259, 128, 128, 256}, {512, 256, 128, 256}, {512, 256, 128, 256}};
  LP L[6][3];
  int p = 1;  // pytree leaf order: alphabetical keys -> W, beta, gamma per layer
  for (int m = 0; m < 6; ++m) {
    int cin = dims[m][0];
    for (int l = 0; l < 3; ++l) {
      LP& lp = L[m][l];
      lp.W = (const float*)d_in[p++];
      lp.beta = (const float*)d_in[p++];
      lp.gamma = (const float*)d_in[p++];
      lp.cin = cin; lp.cout = dims[m][l + 1];
      lp.cinPad = (cin + 31) & ~31;
      cin = lp.cout;
    }
  }

  char* ws = (char*)d_ws; size_t off = 0;
  auto alloc = [&](size_t bytes) -> void* {
    void* r = ws + off; off += (bytes + 255) & ~(size_t)255; return r;
  };
  float* dist = (float*)alloc(2ull * 20000 * 4);
  int*   idx1 = (int*)alloc(2ull * 2048 * 4);
  float* p1 = (float*)alloc(2ull * 2048 * 3 * 4);
  float* p2 = (float*)alloc(2ull * 1024 * 3 * 4);
  float* p3 = (float*)alloc(2ull * 512 * 3 * 4);
  float* p4 = (float*)alloc(2ull * 256 * 3 * 4);
  int*   nidx = (int*)alloc(2ull * 2048 * 64 * 4);
  for (int m = 0; m < 6; ++m)
    for (int l = 0; l < 3; ++l)
      L[m][l].Wh = (_Float16*)alloc((size_t)L[m][l].cinPad * L[m][l].cout * 2);
  _Float16* X  = (_Float16*)alloc(20971520);   // max GEMM input  (ds2: 65536*160*2)
  _Float16* Y0 = (_Float16*)alloc(67108864);   // max layer out   (ds1 l3: 262144*128*2)
  _Float16* Y1 = (_Float16*)alloc(33554432);
  _Float16* f1  = (_Float16*)alloc(2ull * 2048 * 128 * 2);
  _Float16* f2p = (_Float16*)alloc(2ull * 1024 * 256 * 2);
  _Float16* f3  = (_Float16*)alloc(2ull * 512 * 256 * 2);
  _Float16* f4  = (_Float16*)alloc(2ull * 256 * 256 * 2);
  _Float16* i3b = (_Float16*)alloc(2ull * 512 * 256 * 2);
  _Float16* i2b = (_Float16*)alloc(2ull * 1024 * 256 * 2);
  int*   knni = (int*)alloc(2ull * 1024 * 3 * 4);
  float* knnw = (float*)alloc(2ull * 1024 * 3 * 4);

  auto nb = [](long long n, int bs) { return (unsigned)((n + bs - 1) / bs); };

  // 0) pre-pack weights into WMMA B-fragment layout (f16, zero K-padded)
  for (int m = 0; m < 6; ++m)
    for (int l = 0; l < 3; ++l) {
      LP& lp = L[m][l];
      long long n = (long long)lp.cinPad * lp.cout;
      wpack_kernel<<<nb(n, 256), 256, 0, stream>>>(lp.W, lp.Wh, lp.cin, lp.cinPad, lp.cout);
    }

  // 1) FPS (point cloud async-DMA'd into 240KB of the 320KB WGP LDS)
  fps_kernel<<<B_, 1024, 20000 * 3 * sizeof(float), stream>>>(points, dist, idx1, 20000, 2048);
  gather3_kernel<<<nb(2ll * 2048 * 3, 256), 256, 0, stream>>>(points, idx1, p1, 20000, 2048);
  slice_kernel<<<nb(2ll * 1024 * 3, 256), 256, 0, stream>>>(p1, p2, 2048 * 3, 1024 * 3);
  slice_kernel<<<nb(2ll * 512 * 3, 256), 256, 0, stream>>>(p1, p3, 2048 * 3, 512 * 3);
  slice_kernel<<<nb(2ll * 256 * 3, 256), 256, 0, stream>>>(p1, p4, 2048 * 3, 256 * 3);

  auto run_mlp = [&](LP* Ls, const _Float16* Xin, int rows) {
    const _Float16* in = Xin;
    _Float16* outs[3] = {Y0, Y1, Y0};
    for (int l = 0; l < 3; ++l) {
      long long waves = (long long)(rows / 16) * (Ls[l].cout / 64);
      wmma_mlp_kernel<<<nb(waves, 8), 256, 0, stream>>>(
          in, Ls[l].Wh, Ls[l].gamma, Ls[l].beta, outs[l], rows, Ls[l].cinPad, Ls[l].cout);
      in = outs[l];
    }
  };

  // 2) ds1: pts=points(20000), refs=p1(2048), K=64, r=0.2
  ball_query_kernel<<<nb(2ll * 2048, 8), 256, 0, stream>>>(points, p1, nidx, 20000, 2048, 64, 0.04f);
  gather_concat_kernel<<<nb(2ll * 2048 * 64, 256), 256, 0, stream>>>(
      points, (const _Float16*)nullptr, p1, nidx, X, 20000, 2048, 64, 0, 32, 5.0f, 2ll * 2048 * 64);
  run_mlp(L[0], X, 2 * 2048 * 64);
  maxpool_kernel<<<nb(2ll * 2048 * 128, 256), 256, 0, stream>>>(Y0, f1, 64, 128, 2ll * 2048);

  // 3) ds2: pts=p1(2048), refs=p2(1024), K=32, r=0.4, feat C=128
  ball_query_kernel<<<nb(2ll * 1024, 8), 256, 0, stream>>>(p1, p2, nidx, 2048, 1024, 32, 0.16f);
  gather_concat_kernel<<<nb(2ll * 1024 * 32, 256), 256, 0, stream>>>(
      p1, f1, p2, nidx, X, 2048, 1024, 32, 128, 160, 2.5f, 2ll * 1024 * 32);
  run_mlp(L[1], X, 2 * 1024 * 32);
  maxpool_kernel<<<nb(2ll * 1024 * 256, 256), 256, 0, stream>>>(Y0, f2p, 32, 256, 2ll * 1024);

  // 4) ds3: pts=p2(1024), refs=p3(512), K=16, r=0.8, feat C=256
  ball_query_kernel<<<nb(2ll * 512, 8), 256, 0, stream>>>(p2, p3, nidx, 1024, 512, 16, 0.64f);
  gather_concat_kernel<<<nb(2ll * 512 * 16, 256), 256, 0, stream>>>(
      p2, f2p, p3, nidx, X, 1024, 512, 16, 256, 288, 1.25f, 2ll * 512 * 16);
  run_mlp(L[2], X, 2 * 512 * 16);
  maxpool_kernel<<<nb(2ll * 512 * 256, 256), 256, 0, stream>>>(Y0, f3, 16, 256, 2ll * 512);

  // 5) ds4: pts=p3(512), refs=p4(256), K=16, r=1.2, feat C=256
  ball_query_kernel<<<nb(2ll * 256, 8), 256, 0, stream>>>(p3, p4, nidx, 512, 256, 16, 1.44f);
  gather_concat_kernel<<<nb(2ll * 256 * 16, 256), 256, 0, stream>>>(
      p3, f3, p4, nidx, X, 512, 256, 16, 256, 288, 1.0f / 1.2f, 2ll * 256 * 16);
  run_mlp(L[3], X, 2 * 256 * 16);
  maxpool_kernel<<<nb(2ll * 256 * 256, 256), 256, 0, stream>>>(Y0, f4, 16, 256, 2ll * 256);

  // 6) us1: interpolate f4 (p4->p3), concat with f3, MLP
  knn3_kernel<<<nb(2ll * 512, 256), 256, 0, stream>>>(p3, p4, knni, knnw, 512, 256, 2 * 512);
  interp_kernel<<<nb(2ll * 512 * 256, 256), 256, 0, stream>>>(f4, knni, knnw, i3b, 512, 256, 256);
  concat_kernel<<<nb(2ll * 512 * 512, 256), 256, 0, stream>>>(f3, i3b, X, 256, 256, 2ll * 512);
  run_mlp(L[4], X, 2 * 512);                    // f3' lives in Y0 (B*512 x 256)

  // 7) us2: interpolate f3' (p3->p2), concat with f2p, MLP
  knn3_kernel<<<nb(2ll * 1024, 256), 256, 0, stream>>>(p2, p3, knni, knnw, 1024, 512, 2 * 1024);
  interp_kernel<<<nb(2ll * 1024 * 256, 256), 256, 0, stream>>>(Y0, knni, knnw, i2b, 1024, 512, 256);
  concat_kernel<<<nb(2ll * 1024 * 512, 256), 256, 0, stream>>>(f2p, i2b, X, 256, 256, 2ll * 1024);
  run_mlp(L[5], X, 2 * 1024);                   // final f2 lives in Y0 (B*1024 x 256)

  // 8) outputs: (idx2 as float, p2, f2)
  emit_kernel<<<nb(2ll * 1024 * 260, 256), 256, 0, stream>>>(
      idx1, p1, Y0, (float*)d_out, 1024, 2048, 256, 2 * 1024 * 260);
}
